// PostInferer_68264210203186
// MI455X (gfx1250) — compile-verified, matrix operand order checked by reading, not run on hardware
//
#include <hip/hip_runtime.h>
#include <math.h>

#define D     256
#define NB    16
#define NTOK  1025
#define NSAT  1024
#define NH    8
#define DH    32

typedef __attribute__((ext_vector_type(16))) __bf16 v16bf;
typedef __attribute__((ext_vector_type(8)))  float  v8f;

__device__ __forceinline__ int koff(int v, int g) {
    // K-dim element pair base for 16-bit A/B WMMA operand, VGPR v, lane-half g
    return (v < 4) ? (2 * v + 8 * g) : (16 + 2 * (v - 4) + 8 * g);
}

__device__ __forceinline__ v8f wmma_bf16(v16bf a, v16bf b, v8f c) {
    return __builtin_amdgcn_wmma_f32_16x16x32_bf16(false, a, false, b, (short)0, c,
                                                   false, false);
}

// ---------------------------------------------------------------------------
// Pre-pack one 256x256 f32 weight matrix into bf16 WMMA B-operand layout:
// tile (kk, nt) -> 32 lanes x 16 bf16 (32 bytes/lane, loadable as 2x b128).
// ---------------------------------------------------------------------------
__global__ void pack_w(const float* __restrict__ W, __bf16* __restrict__ P) {
    int wave = (int)((blockIdx.x * blockDim.x + threadIdx.x) >> 5);
    if (wave >= 128) return;                 // 8 kk x 16 nt
    int kk = wave >> 4, nt = wave & 15;
    int lane = threadIdx.x & 31;
    int lr = lane & 15, g = lane >> 4;
    int ncol = nt * 16 + lr;
    v16bf t;
    #pragma unroll
    for (int v = 0; v < 8; ++v) {
        int k0 = kk * 32 + koff(v, g);
        t[2 * v]     = (__bf16)W[(size_t)k0 * D + ncol];
        t[2 * v + 1] = (__bf16)W[(size_t)(k0 + 1) * D + ncol];
    }
    *(v16bf*)(P + ((size_t)(kk * 16 + nt) * 32 + lane) * 16) = t;
}

// ---------------------------------------------------------------------------
// Pack K (B x Lk x D, f32) into per-(b, key-tile, head) WMMA A-layout tiles.
// Tile index = (b*ktiles + kt)*NH + h ; 512 bf16 per tile.
// ---------------------------------------------------------------------------
__global__ void pack_k(const float* __restrict__ K, __bf16* __restrict__ P,
                       int Lk, int ktiles) {
    int wave = (int)((blockIdx.x * blockDim.x + threadIdx.x) >> 5);
    int total = NB * ktiles * NH;
    if (wave >= total) return;
    int h  = wave % NH;
    int kt = (wave / NH) % ktiles;
    int b  = wave / (NH * ktiles);
    int lane = threadIdx.x & 31;
    int lr = lane & 15, g = lane >> 4;
    int key = kt * 16 + lr; if (key >= Lk) key = Lk - 1;
    const float* kr = K + ((size_t)b * Lk + key) * D + h * DH;
    v16bf t;
    #pragma unroll
    for (int v = 0; v < 8; ++v) {
        float2 x = *(const float2*)(kr + koff(v, g));
        t[2 * v] = (__bf16)x.x; t[2 * v + 1] = (__bf16)x.y;
    }
    *(v16bf*)(P + ((size_t)wave * 32 + lane) * 16) = t;
}

// ---------------------------------------------------------------------------
// Pack V (B x Lk x D, f32) into per-(b, 32-key block, head, d-tile) WMMA
// B-layout tiles. Tile index = ((b*kbs + kb)*NH + h)*2 + dt ; 512 bf16/tile.
// ---------------------------------------------------------------------------
__global__ void pack_v(const float* __restrict__ V, __bf16* __restrict__ P,
                       int Lk, int kbs) {
    int wave = (int)((blockIdx.x * blockDim.x + threadIdx.x) >> 5);
    int total = NB * kbs * NH * 2;
    if (wave >= total) return;
    int dt = wave & 1;
    int rest = wave >> 1;
    int h  = rest % NH;
    int kb = (rest / NH) % kbs;
    int b  = rest / (NH * kbs);
    int lane = threadIdx.x & 31;
    int lr = lane & 15, g = lane >> 4;
    int dcol = h * DH + dt * 16 + lr;
    const float* vb = V + (size_t)b * Lk * D;
    v16bf t;
    #pragma unroll
    for (int v = 0; v < 8; ++v) {
        int k0 = kb * 32 + koff(v, g);
        int k0c = k0 < Lk ? k0 : Lk - 1;
        int k1c = (k0 + 1) < Lk ? (k0 + 1) : Lk - 1;
        t[2 * v]     = (__bf16)vb[(size_t)k0c * D + dcol];
        t[2 * v + 1] = (__bf16)vb[(size_t)k1c * D + dcol];
    }
    *(v16bf*)(P + ((size_t)wave * 32 + lane) * 16) = t;
}

// ---------------------------------------------------------------------------
// Y[b*yT + yOff + m, n] = sum_k X[b*xT + xOff + m, k] * W[k, n]
// Wp is pre-packed B-layout. One wave computes a 16x64 strip (4 n-tiles).
// ---------------------------------------------------------------------------
__global__ void gemm_proj(const float* __restrict__ X, const __bf16* __restrict__ Wp,
                          float* __restrict__ Y,
                          int rows, int xT, int xOff, int yT, int yOff, int mtiles) {
    int wave = (int)((blockIdx.x * blockDim.x + threadIdx.x) >> 5);
    int total = NB * mtiles * 4;
    if (wave >= total) return;
    int ng = wave % 4;
    int mt = (wave / 4) % mtiles;
    int b  = wave / (4 * mtiles);
    int lane = threadIdx.x & 31;
    int lr = lane & 15, g = lane >> 4;

    int arow = mt * 16 + lr; if (arow >= rows) arow = rows - 1;
    const float* xrow = X + (size_t)(b * xT + xOff + arow) * D;

    v8f c0 = {}, c1 = {}, c2 = {}, c3 = {};
    #pragma unroll
    for (int kk = 0; kk < 8; ++kk) {
        v16bf a;
        #pragma unroll
        for (int v = 0; v < 8; ++v) {
            int k0 = kk * 32 + koff(v, g);
            float2 xv = *(const float2*)(xrow + k0);
            a[2 * v] = (__bf16)xv.x; a[2 * v + 1] = (__bf16)xv.y;
        }
        const __bf16* wt = Wp + ((size_t)(kk * 16 + ng * 4) * 32 + lane) * 16;
        v16bf b0 = *(const v16bf*)(wt);
        v16bf b1 = *(const v16bf*)(wt + 32 * 16);
        v16bf b2 = *(const v16bf*)(wt + 64 * 16);
        v16bf b3 = *(const v16bf*)(wt + 96 * 16);
        c0 = wmma_bf16(a, b0, c0);
        c1 = wmma_bf16(a, b1, c1);
        c2 = wmma_bf16(a, b2, c2);
        c3 = wmma_bf16(a, b3, c3);
    }
    #pragma unroll
    for (int r = 0; r < 8; ++r) {
        int m = mt * 16 + r + 8 * g;
        if (m < rows) {
            float* yrow = Y + (size_t)(b * yT + yOff + m) * D + ng * 64 + lr;
            yrow[0]  = c0[r];
            yrow[16] = c1[r];
            yrow[32] = c2[r];
            yrow[48] = c3[r];
        }
    }
}

// ---------------------------------------------------------------------------
// out[b, 1+m, d] = sum_n graph[b, n, m] * gvin[b, 1+n, d]
// One wave per 16x64 strip: each graph A-tile feeds 4 WMMAs.
// ---------------------------------------------------------------------------
__global__ void graph_mix(const float* __restrict__ gvin, const float* __restrict__ graph,
                          float* __restrict__ out) {
    const int mtiles = NSAT / 16;
    int wave = (int)((blockIdx.x * blockDim.x + threadIdx.x) >> 5);
    int total = NB * mtiles * 4;
    if (wave >= total) return;
    int ng = wave % 4;
    int mt = (wave / 4) % mtiles;
    int b  = wave / (4 * mtiles);
    int lane = threadIdx.x & 31;
    int lr = lane & 15, g = lane >> 4;

    int mcol = mt * 16 + lr;
    const float* gb = graph + (size_t)b * NSAT * NSAT;
    const float* sb = gvin + ((size_t)b * NTOK + 1) * D;

    v8f c0 = {}, c1 = {}, c2 = {}, c3 = {};
    for (int kb = 0; kb < NSAT / 32; ++kb) {
        v16bf a;
        #pragma unroll
        for (int v = 0; v < 8; ++v) {
            int n0 = kb * 32 + koff(v, g);
            a[2 * v]     = (__bf16)gb[(size_t)n0 * NSAT + mcol];
            a[2 * v + 1] = (__bf16)gb[(size_t)(n0 + 1) * NSAT + mcol];
        }
        #pragma unroll
        for (int j = 0; j < 4; ++j) {
            int dcol = (ng * 4 + j) * 16 + lr;
            v16bf bm;
            #pragma unroll
            for (int v = 0; v < 8; ++v) {
                int n0 = kb * 32 + koff(v, g);
                bm[2 * v]     = (__bf16)sb[(size_t)n0 * D + dcol];
                bm[2 * v + 1] = (__bf16)sb[(size_t)(n0 + 1) * D + dcol];
            }
            if (j == 0) c0 = wmma_bf16(a, bm, c0);
            else if (j == 1) c1 = wmma_bf16(a, bm, c1);
            else if (j == 2) c2 = wmma_bf16(a, bm, c2);
            else c3 = wmma_bf16(a, bm, c3);
        }
    }
    #pragma unroll
    for (int r = 0; r < 8; ++r) {
        int m = mt * 16 + r + 8 * g;
        float* yrow = out + ((size_t)b * NTOK + 1 + m) * D + ng * 64 + lr;
        yrow[0]  = c0[r];
        yrow[16] = c1[r];
        yrow[32] = c2[r];
        yrow[48] = c3[r];
    }
}

// ---------------------------------------------------------------------------
// Flash attention over pre-packed K/V. One wave per (b, h, q-tile of 16).
// Scores computed transposed (A = K-tile, B = Q^T): softmax reductions are
// per-lane and the P-tile lands directly in bf16 A-matrix layout for P@V.
// ---------------------------------------------------------------------------
__global__ void attn(const float* __restrict__ Q, const __bf16* __restrict__ pK,
                     const __bf16* __restrict__ pV, float* __restrict__ O,
                     int Lq, int Lk, int qtiles) {
    int wave = (int)((blockIdx.x * blockDim.x + threadIdx.x) >> 5);
    int total = NB * NH * qtiles;
    if (wave >= total) return;
    int qt = wave % qtiles;
    int h  = (wave / qtiles) % NH;
    int b  = wave / (qtiles * NH);
    int lane = threadIdx.x & 31;
    int lr = lane & 15, g = lane >> 4;
    const float scale = 0.17677669529663687f;  // 1/sqrt(32)
    const float NEG = -1e30f;

    int nkb = (Lk + 31) / 32;
    int ktiles = 2 * nkb;

    const __bf16* kbase = pK + (((size_t)b * ktiles) * NH + h) * 512 + lane * 16;
    const __bf16* vbase = pV + (((size_t)b * nkb) * NH + h) * 2 * 512 + lane * 16;
    const size_t kstep = (size_t)NH * 512;
    const size_t vstep = (size_t)NH * 2 * 512;

    // B operand: lane = query column
    const float* Qb = Q + (size_t)b * Lq * D + h * DH;
    int qrow = qt * 16 + lr; if (qrow >= Lq) qrow = Lq - 1;
    v16bf bq;
    #pragma unroll
    for (int v = 0; v < 8; ++v) {
        float2 qv = *(const float2*)(Qb + (size_t)qrow * D + koff(v, g));
        bq[2 * v] = (__bf16)qv.x; bq[2 * v + 1] = (__bf16)qv.y;
    }

    // ---- pass 1: per-half-lane running max & sum-of-exp over this lane's keys
    float mRun = NEG, sRun = 0.f;
    for (int kb = 0; kb < nkb; ++kb) {
        v16bf a0 = *(const v16bf*)(kbase + (size_t)(2 * kb) * kstep);
        v16bf a1 = *(const v16bf*)(kbase + (size_t)(2 * kb + 1) * kstep);
        v8f c0 = {}, c1 = {};
        c0 = wmma_bf16(a0, bq, c0);
        c1 = wmma_bf16(a1, bq, c1);
        float s[16], tmax = NEG;
        #pragma unroll
        for (int r = 0; r < 8; ++r) {
            int key0 = kb * 32 + r + 8 * g;
            int key1 = key0 + 16;
            s[r]     = (key0 < Lk) ? c0[r] * scale : NEG;
            s[8 + r] = (key1 < Lk) ? c1[r] * scale : NEG;
            tmax = fmaxf(tmax, fmaxf(s[r], s[8 + r]));
        }
        float mNew = fmaxf(mRun, tmax);
        sRun *= __expf(mRun - mNew);
        #pragma unroll
        for (int i = 0; i < 16; ++i) sRun += __expf(s[i] - mNew);
        mRun = mNew;
    }
    // merge the two lane halves (each saw a disjoint key subset)
    float mO = __shfl_xor(mRun, 16, 32);
    float sO = __shfl_xor(sRun, 16, 32);
    float Mf = fmaxf(mRun, mO);
    float Sf = sRun * __expf(mRun - Mf) + sO * __expf(mO - Mf);
    float invS = 1.0f / Sf;

    // ---- pass 2: P = exp(S - M), O += P @ V
    v8f accA = {}, accB = {};
    for (int kb = 0; kb < nkb; ++kb) {
        v16bf a0 = *(const v16bf*)(kbase + (size_t)(2 * kb) * kstep);
        v16bf a1 = *(const v16bf*)(kbase + (size_t)(2 * kb + 1) * kstep);
        v8f c0 = {}, c1 = {};
        c0 = wmma_bf16(a0, bq, c0);
        c1 = wmma_bf16(a1, bq, c1);
        float p0[8], p1[8];
        #pragma unroll
        for (int r = 0; r < 8; ++r) {
            int key0 = kb * 32 + r + 8 * g;
            int key1 = key0 + 16;
            p0[r] = (key0 < Lk) ? __expf(c0[r] * scale - Mf) : 0.f;
            p1[r] = (key1 < Lk) ? __expf(c1[r] * scale - Mf) : 0.f;
        }
        // P-tile is already in A-matrix layout: lane = query, key halves by g
        v16bf ap;
        #pragma unroll
        for (int v = 0; v < 4; ++v) {
            ap[2 * v] = (__bf16)p0[2 * v]; ap[2 * v + 1] = (__bf16)p0[2 * v + 1];
        }
        #pragma unroll
        for (int v = 4; v < 8; ++v) {
            ap[2 * v] = (__bf16)p1[2 * (v - 4)]; ap[2 * v + 1] = (__bf16)p1[2 * (v - 4) + 1];
        }
        v16bf bv0 = *(const v16bf*)(vbase + (size_t)kb * vstep);
        v16bf bv1 = *(const v16bf*)(vbase + (size_t)kb * vstep + 512);
        accA = wmma_bf16(ap, bv0, accA);
        accB = wmma_bf16(ap, bv1, accB);
    }

    float* Ob = O + (size_t)b * Lq * D + h * DH;
    int iS = __float_as_int(invS);
    #pragma unroll
    for (int r = 0; r < 8; ++r) {
        int qm = r + 8 * g;  // query index of this accumulator row
        float inv = __int_as_float(__builtin_amdgcn_ds_bpermute(qm << 2, iS));
        int qg = qt * 16 + qm;
        if (qg < Lq) {
            Ob[(size_t)qg * D + lr]      = accA[r] * inv;
            Ob[(size_t)qg * D + 16 + lr] = accB[r] * inv;
        }
    }
}

// ---------------------------------------------------------------------------
// Y = LayerNorm(T + Xres) * gamma + beta ; one wave per row (D=256 -> 8/lane)
// ---------------------------------------------------------------------------
__global__ void resid_ln(const float* __restrict__ T, const float* __restrict__ Xres,
                         const float* __restrict__ gam, const float* __restrict__ bet,
                         float* __restrict__ Y,
                         int Lq, int xT, int xOff, int yT, int yOff) {
    int row = (int)((blockIdx.x * blockDim.x + threadIdx.x) >> 5);
    int total = NB * Lq;
    if (row >= total) return;
    int b = row / Lq, r = row % Lq;
    int lane = threadIdx.x & 31;
    const float* t = T + (size_t)row * D;
    const float* x = Xres + (size_t)(b * xT + xOff + r) * D;
    float v[8], sum = 0.f, sq = 0.f;
    #pragma unroll
    for (int i = 0; i < 8; ++i) {
        float a = t[lane + 32 * i] + x[lane + 32 * i];
        v[i] = a; sum += a; sq += a * a;
    }
    #pragma unroll
    for (int off = 16; off > 0; off >>= 1) {
        sum += __shfl_xor(sum, off, 32);
        sq  += __shfl_xor(sq,  off, 32);
    }
    float mu  = sum * (1.0f / D);
    float var = sq  * (1.0f / D) - mu * mu;
    float rs  = rsqrtf(var + 1e-5f);
    float* y = Y + (size_t)(b * yT + yOff + r) * D;
    #pragma unroll
    for (int i = 0; i < 8; ++i) {
        int c = lane + 32 * i;
        y[c] = (v[i] - mu) * rs * gam[c] + bet[c];
    }
}

__global__ void copy_sat(const float* __restrict__ gv, float* __restrict__ dst) {
    size_t i = (size_t)blockIdx.x * blockDim.x + threadIdx.x;
    const size_t per = (size_t)NSAT * D;
    if (i >= (size_t)NB * per) return;
    size_t b = i / per, r = i - b * per;
    dst[i] = gv[b * (size_t)NTOK * D + D + r];
}

// ---------------------------------------------------------------------------
extern "C" void kernel_launch(void* const* d_in, const int* in_sizes, int n_in,
                              void* d_out, int out_size, void* d_ws, size_t ws_size,
                              hipStream_t stream) {
    (void)in_sizes; (void)n_in; (void)out_size; (void)ws_size;
    const float* gv0   = (const float*)d_in[0];
    const float* graph = (const float*)d_in[1];
    const float* Wq    = (const float*)d_in[2];
    const float* Wk    = (const float*)d_in[3];
    const float* Wv    = (const float*)d_in[4];
    const float* Wo    = (const float*)d_in[5];
    const float* lg    = (const float*)d_in[6];
    const float* lb    = (const float*)d_in[7];
    float* out = (float*)d_out;

    const size_t S = (size_t)NB * NTOK * D;
    float* bufA = (float*)d_ws;
    float* bufB = bufA + S;
    float* bufQ = bufB + S;
    float* bufK = bufQ + S;
    float* bufV = bufK + S;
    float* bufO = bufV + S;
    float* bufP = bufQ;  // out-projection scratch (Q is dead by then)

    const int DD = D * D;
    const size_t PW = (size_t)D * D;              // bf16 elems per packed matrix
    __bf16* pw = (__bf16*)(bufO + S);             // 16 matrices * 128KB = 2MB
    __bf16* pQ = pw;
    __bf16* pK = pw + 4 * PW;
    __bf16* pV = pw + 8 * PW;
    __bf16* pO = pw + 12 * PW;
    __bf16* pKa = pw + 16 * PW;                   // packed K tiles (<= 8.7MB)
    const size_t PKA = (size_t)NB * 66 * NH * 512;
    __bf16* pVb = pKa + PKA;                      // packed V tiles (<= 8.7MB)

    auto grid4 = [](int waves) { return (waves + 3) / 4; };  // 4 waves/block

    // ================= pre-pack all weights into WMMA B-layout ===============
    for (int i = 0; i < 4; ++i) {
        pack_w<<<32, 128, 0, stream>>>(Wq + (size_t)i * DD, pQ + i * PW);
        pack_w<<<32, 128, 0, stream>>>(Wk + (size_t)i * DD, pK + i * PW);
        pack_w<<<32, 128, 0, stream>>>(Wv + (size_t)i * DD, pV + i * PW);
        pack_w<<<32, 128, 0, stream>>>(Wo + (size_t)i * DD, pO + i * PW);
    }

    // ================= layer 0: relay <- cross-attn(relay, sat) =============
    gemm_proj<<<grid4(NB * 1 * 4),  128, 0, stream>>>(gv0, pQ, bufQ, 1,    NTOK, 0, 1,    0, 1);
    gemm_proj<<<grid4(NB * 64 * 4), 128, 0, stream>>>(gv0, pK, bufK, 1024, NTOK, 1, 1024, 0, 64);
    gemm_proj<<<grid4(NB * 64 * 4), 128, 0, stream>>>(gv0, pV, bufV, 1024, NTOK, 1, 1024, 0, 64);
    pack_k<<<grid4(NB * 64 * NH),     128, 0, stream>>>(bufK, pKa, 1024, 64);
    pack_v<<<grid4(NB * 32 * NH * 2), 128, 0, stream>>>(bufV, pVb, 1024, 32);
    attn<<<grid4(NB * NH * 1), 128, 0, stream>>>(bufQ, pKa, pVb, bufO, 1, 1024, 1);
    gemm_proj<<<grid4(NB * 1 * 4), 128, 0, stream>>>(bufO, pO, bufP, 1, 1, 0, 1, 0, 1);
    resid_ln<<<grid4(NB * 1), 128, 0, stream>>>(bufP, gv0, lg, lb, bufA, 1, NTOK, 0, NTOK, 0);

    // ================= graph mixing: sat rows of bufA ========================
    graph_mix<<<grid4(NB * 64 * 4), 128, 0, stream>>>(gv0, graph, bufA);

    // ================= layers 1..2: full self-attention ======================
    float* cur = bufA;
    float* nxt = bufB;
    for (int i = 1; i <= 2; ++i) {
        const int mt = 65;  // ceil(1025/16)
        gemm_proj<<<grid4(NB * mt * 4), 128, 0, stream>>>(cur, pQ + i * PW, bufQ, NTOK, NTOK, 0, NTOK, 0, mt);
        gemm_proj<<<grid4(NB * mt * 4), 128, 0, stream>>>(cur, pK + i * PW, bufK, NTOK, NTOK, 0, NTOK, 0, mt);
        gemm_proj<<<grid4(NB * mt * 4), 128, 0, stream>>>(cur, pV + i * PW, bufV, NTOK, NTOK, 0, NTOK, 0, mt);
        pack_k<<<grid4(NB * 66 * NH),     128, 0, stream>>>(bufK, pKa, NTOK, 66);
        pack_v<<<grid4(NB * 33 * NH * 2), 128, 0, stream>>>(bufV, pVb, NTOK, 33);
        attn<<<grid4(NB * NH * mt), 128, 0, stream>>>(bufQ, pKa, pVb, bufO, NTOK, NTOK, mt);
        gemm_proj<<<grid4(NB * mt * 4), 128, 0, stream>>>(bufO, pO + i * PW, bufP, NTOK, NTOK, 0, NTOK, 0, mt);
        resid_ln<<<grid4(NB * NTOK), 128, 0, stream>>>(bufP, cur, lg + i * D, lb + i * D,
                                                       nxt, NTOK, NTOK, 0, NTOK, 0);
        float* t = cur; cur = nxt; nxt = t;
    }
    // cur == bufA holds post-layer-2 gv

    // ================= outputs: sat_u then final cross-attn z ===============
    {
        size_t n = (size_t)NB * NSAT * D;
        copy_sat<<<(unsigned)((n + 255) / 256), 256, 0, stream>>>(cur, out + (size_t)NB * D);
    }
    gemm_proj<<<grid4(NB * 1 * 4),  128, 0, stream>>>(cur, pQ + 3 * PW, bufQ, 1,    NTOK, 0, 1,    0, 1);
    gemm_proj<<<grid4(NB * 64 * 4), 128, 0, stream>>>(cur, pK + 3 * PW, bufK, 1024, NTOK, 1, 1024, 0, 64);
    gemm_proj<<<grid4(NB * 64 * 4), 128, 0, stream>>>(cur, pV + 3 * PW, bufV, 1024, NTOK, 1, 1024, 0, 64);
    pack_k<<<grid4(NB * 64 * NH),     128, 0, stream>>>(bufK, pKa, 1024, 64);
    pack_v<<<grid4(NB * 32 * NH * 2), 128, 0, stream>>>(bufV, pVb, 1024, 32);
    attn<<<grid4(NB * NH * 1), 128, 0, stream>>>(bufQ, pKa, pVb, bufO, 1, 1024, 1);
    gemm_proj<<<grid4(NB * 1 * 4), 128, 0, stream>>>(bufO, pO + 3 * PW, bufP, 1, 1, 0, 1, 0, 1);
    resid_ln<<<grid4(NB * 1), 128, 0, stream>>>(bufP, cur, lg + 3 * D, lb + 3 * D,
                                                out, 1, NTOK, 0, 1, 0);
}
